// LoRAXSLinear_76733885710753
// MI455X (gfx1250) — compile-verified, matrix operand order checked by reading, not run on hardware
//
#include <hip/hip_runtime.h>

// ---------- types ----------
typedef __attribute__((ext_vector_type(16))) __bf16        v16bf;
typedef __attribute__((ext_vector_type(8)))  float         v8f;
typedef __attribute__((ext_vector_type(4)))  unsigned int  v4u;

union FragBF16 {
    v16bf v;
    v4u   q[2];
};

// round-to-nearest-even f32 -> bf16 (bit pattern)
__device__ __forceinline__ unsigned short f32_to_bf16(float f) {
    unsigned int u = __builtin_bit_cast(unsigned int, f);
    u += 0x7FFFu + ((u >> 16) & 1u);
    return (unsigned short)(u >> 16);
}

// ---------- problem constants (from reference) ----------
#define MB    8192      // B*S = 4*2048
#define NDIM  2048      // D_OUT
#define KDIM  2048      // D_IN
#define RANK  64

// ---------- tiny chain: SR = sigma @ R  (64x64) ----------
__global__ void sr_kernel(const float* __restrict__ sigma,
                          const float* __restrict__ Rm,
                          float* __restrict__ SR) {
    int o = blockIdx.x;        // 0..63
    int i = threadIdx.x;       // 0..63
    float acc = 0.f;
    #pragma unroll 8
    for (int k = 0; k < RANK; ++k)
        acc += sigma[o * RANK + k] * Rm[k * RANK + i];
    SR[o * RANK + i] = acc;
}

// ---------- USR = U @ SR  (2048 x 64) ----------
__global__ void usr_kernel(const float* __restrict__ U,
                           const float* __restrict__ SR,
                           float* __restrict__ USR) {
    int idx = blockIdx.x * blockDim.x + threadIdx.x;   // 2048*64
    int o = idx >> 6;
    int r = idx & 63;
    float acc = 0.f;
    #pragma unroll 8
    for (int k = 0; k < RANK; ++k)
        acc += U[o * RANK + k] * SR[k * RANK + r];
    USR[idx] = acc;
}

// ---------- W_eff(bf16) = bf16(W + USR @ Vt)  (2048 x 2048) ----------
__global__ __launch_bounds__(256)
void weff_kernel(const float* __restrict__ W,
                 const float* __restrict__ USR,
                 const float* __restrict__ Vt,
                 unsigned short* __restrict__ Wb) {
    int idx = blockIdx.x * 256 + threadIdx.x;  // o*2048 + i
    int o = idx >> 11;
    int i = idx & 2047;
    float acc = W[idx];
    #pragma unroll 8
    for (int r = 0; r < RANK; ++r)             // ALPHA == 1.0
        acc += USR[(o << 6) + r] * Vt[(r << 11) + i];
    Wb[idx] = f32_to_bf16(acc);
}

// ---------- x(bf16) cast ----------
__global__ __launch_bounds__(256)
void castx_kernel(const float* __restrict__ x,
                  unsigned short* __restrict__ Xb) {
    int idx = blockIdx.x * 256 + threadIdx.x;  // exactly MB*KDIM threads launched
    Xb[idx] = f32_to_bf16(x[idx]);
}

// ---------- main WMMA GEMM: out = Xb @ Wb^T + bias ----------
// Block = 8 waves. Block computes a 64 x 512 strip of C: one shared 64-row A
// slab (m0) and 512 B rows (nB0..nB0+511), each wave owning a 64-col strip.
// ALL operands are staged into LDS via async global->LDS DMA (ASYNCcnt),
// double-buffered in 32-k stages. WMMA consumes fragments from LDS via
// ds_load_b128 with one B fragment live at a time to keep VGPRs < 256.
__global__ __launch_bounds__(256)
void wmma_gemm_kernel(const unsigned short* __restrict__ Xb,  // MB x KDIM bf16
                      const unsigned short* __restrict__ Wb,  // NDIM x KDIM bf16
                      const float* __restrict__ bias,
                      float* __restrict__ out) {
    // dynamic LDS: Abuf 2 x 2048 ushort (4 KiB each), Bbuf 2 x 16384 ushort
    // (32 KiB each) -> 72 KiB total (gfx1250 WGP has 320 KiB).
    extern __shared__ unsigned short smem[];
    unsigned short* Abuf = smem;            // [2][64*32]
    unsigned short* Bbuf = smem + 2 * 2048; // [2][512*32]

    const int tid  = threadIdx.x;
    const int lane = tid & 31;
    const int wave = tid >> 5;

    const int m0  = (blockIdx.x >> 2) * 64;   // shared by all 8 waves
    const int nB0 = (blockIdx.x & 3) * 512;   // block's B-row window
    const int n0  = nB0 + wave * 64;          // per-wave strip

    const int r16  = lane & 15;               // row-in-tile (A) / col-in-tile (B,C)
    const int half = lane >> 4;               // K-half selector per ISA layout

    v8f c[4][4] = {};                         // 16 fp32 16x16 accum tiles

    // cooperative DMA pattern: thread t covers row t>>2 (0..63), k-chunk (t&3)*8
    const int ldRow = tid >> 2;
    const int ldKof = (tid & 3) * 8;
    const unsigned short* aSrc = Xb + (size_t)(m0 + ldRow) * KDIM + ldKof;
    const unsigned short* bSrc = Wb + (size_t)(nB0 + ldRow) * KDIM + ldKof;
    // ISA aperture rule: low 32 bits of a flat LDS address ARE the LDS byte addr
    const unsigned aDst0 =
        (unsigned)(unsigned long long)(const void*)&Abuf[ldRow * 32 + ldKof];
    const unsigned bDst0 =
        (unsigned)(unsigned long long)(const void*)&Bbuf[ldRow * 32 + ldKof];

    // issue one stage: 1 async op for A (wave moves 512B) + 8 for B q-groups
    #define ISSUE_STAGE(stage)                                                   \
        do {                                                                     \
            const unsigned _sel = (unsigned)((stage) & 1);                       \
            unsigned _ad = aDst0 + _sel * 4096u;                                 \
            const unsigned short* _as = aSrc + (size_t)(stage) * 32;             \
            asm volatile("global_load_async_to_lds_b128 %0, %1, off"             \
                         :: "v"(_ad), "v"(_as) : "memory");                      \
            unsigned _bd = bDst0 + _sel * 32768u;                                \
            const unsigned short* _bs = bSrc + (size_t)(stage) * 32;             \
            _Pragma("unroll")                                                    \
            for (int _q = 0; _q < 8; ++_q) {                                     \
                asm volatile("global_load_async_to_lds_b128 %0, %1, off"         \
                             :: "v"(_bd + (unsigned)_q * 4096u),                 \
                                "v"(_bs + (size_t)_q * 64 * KDIM) : "memory");   \
            }                                                                    \
        } while (0)

    ISSUE_STAGE(0);

    const int STAGES = KDIM / 32;             // 64
    for (int s = 0; s < STAGES; ++s) {
        if (s + 1 < STAGES) {
            ISSUE_STAGE(s + 1);
            asm volatile("s_wait_asynccnt 9" ::: "memory");   // stage s landed
        } else {
            asm volatile("s_wait_asynccnt 0" ::: "memory");
        }
        __syncthreads();                      // all waves' slab writes visible

        const unsigned short* aSlab = Abuf + (s & 1) * 2048;
        const unsigned short* bSlab = Bbuf + (s & 1) * 16384 + wave * (64 * 32);

        // A fragments (resident across the j loop)
        FragBF16 a[4];
        #pragma unroll
        for (int i = 0; i < 4; ++i) {
            const unsigned short* p = aSlab + (16 * i + r16) * 32 + half * 8;
            a[i].q[0] = *(const v4u*)(p);
            a[i].q[1] = *(const v4u*)(p + 16);
        }

        // one B fragment live at a time -> low VGPR pressure
        #pragma unroll
        for (int j = 0; j < 4; ++j) {
            FragBF16 b;
            const unsigned short* p = bSlab + (16 * j + r16) * 32 + half * 8;
            b.q[0] = *(const v4u*)(p);
            b.q[1] = *(const v4u*)(p + 16);
            #pragma unroll
            for (int i = 0; i < 4; ++i)
                c[i][j] = __builtin_amdgcn_wmma_f32_16x16x32_bf16(
                    false, a[i].v, false, b.v,
                    (short)0, c[i][j], false, false);
        }

        __syncthreads();   // reads of buf[s&1] drained before stage s+2 overwrites
    }
    #undef ISSUE_STAGE

    // epilogue: bias add + store per documented C/D layout
    // c[i][j][r] holds C[m0+16i + r + 8*half][n0+16j + r16]
    #pragma unroll
    for (int j = 0; j < 4; ++j) {
        const int n = n0 + 16 * j + r16;
        const float bv = bias[n];
        #pragma unroll
        for (int i = 0; i < 4; ++i) {
            const int mBase = m0 + 16 * i + 8 * half;
            #pragma unroll
            for (int r = 0; r < 8; ++r) {
                out[(size_t)(mBase + r) * NDIM + n] = c[i][j][r] + bv;
            }
        }
    }
}

// ---------- launch ----------
extern "C" void kernel_launch(void* const* d_in, const int* in_sizes, int n_in,
                              void* d_out, int out_size, void* d_ws, size_t ws_size,
                              hipStream_t stream) {
    const float* x      = (const float*)d_in[0];
    const float* weight = (const float*)d_in[1];
    const float* bias   = (const float*)d_in[2];
    const float* U      = (const float*)d_in[3];
    const float* sigma  = (const float*)d_in[4];
    const float* Rm     = (const float*)d_in[5];
    const float* Vt     = (const float*)d_in[6];
    float* out = (float*)d_out;

    // workspace layout (all 16B-aligned):
    //   SR   : 64*64 f32        @ 0        (16 KiB)
    //   USR  : 2048*64 f32      @ 16384    (512 KiB)
    //   Wb   : 2048*2048 bf16   @ 540672   (8 MiB)
    //   Xb   : 8192*2048 bf16   @ 8929280  (32 MiB)
    char* ws = (char*)d_ws;
    float*          SR  = (float*)(ws);
    float*          USR = (float*)(ws + 16384);
    unsigned short* Wb  = (unsigned short*)(ws + 540672);
    unsigned short* Xb  = (unsigned short*)(ws + 8929280);

    sr_kernel  <<<64,   64,  0, stream>>>(sigma, Rm, SR);
    usr_kernel <<<512,  256, 0, stream>>>(U, SR, USR);            // 2048*64/256
    weff_kernel<<<16384,256, 0, stream>>>(weight, USR, Vt, Wb);   // 2048*2048/256
    castx_kernel<<<65536,256,0, stream>>>(x, Xb);                 // 8192*2048/256

    // 128 m-blocks x 4 n-blocks = 512 blocks of 8 waves; 72 KiB dynamic LDS
    wmma_gemm_kernel<<<512, 256, 73728, stream>>>(Xb, Wb, bias, out);
}
